// MiniGPT_10668698763691
// MI455X (gfx1250) — matrix operand order repaired and checked by reading, not compiled, not run on hardware
//
#include <hip/hip_runtime.h>
#include <hip/hip_bf16.h>

// ---------------------------------------------------------------------------
// MiniGPT forward on gfx1250 (MI455X): all GEMMs + attention on
// v_wmma_f32_16x16x32_f16 (f32 accumulate), flash-style causal attention.
// Double-buffered LDS staging (1 barrier/chunk, loads overlap WMMA),
// vector-only DS traffic, template-specialized epilogues (no per-element
// branches), global_prefetch on the activation stream.
// ---------------------------------------------------------------------------

typedef __attribute__((ext_vector_type(16))) _Float16 v16h;
typedef __attribute__((ext_vector_type(8)))  _Float16 v8h;
typedef __attribute__((ext_vector_type(8)))  float    v8f;

union Frag16 { v16h v; v8h h2[2]; _Float16 e[16]; };
union Frag8H { v8h  v; _Float16 e[8]; };
union AccF8  { v8f  v; float e[8]; };

#define L_n 6
#define H_n 16
#define C_n 1024
#define HS_n 64
#define T_n 1024
#define B_n 2
#define V_n 32000
#define M_n (B_n * T_n)   // 2048 rows of activations

static __device__ __forceinline__ v8f wmma16(v16h a, v16h b, v8f c) {
  // D = A(16x32 f16) * B(32x16 f16) + C(16x16 f32)
  return __builtin_amdgcn_wmma_f32_16x16x32_f16(false, a, false, b,
                                                (short)0, c, false, false);
}

// ---------------------------------------------------------------------------
// Embedding: x[b,t,:] = tok_emb[idx[b,t],:] + pos_emb[t,:]
// ---------------------------------------------------------------------------
__global__ __launch_bounds__(256) void embed_kernel(
    const int* __restrict__ idx, const float* __restrict__ tok,
    const float* __restrict__ pos, float* __restrict__ x) {
  const int row = blockIdx.x;            // b*T + t
  const int t   = threadIdx.x;
  const int id  = idx[row];
  const int tp  = row % T_n;
  const float* te = tok + (size_t)id * C_n;
  const float* pe = pos + (size_t)tp * C_n;
  float* xr = x + (size_t)row * C_n;
#pragma unroll
  for (int i = 0; i < 4; ++i) {
    const int c = t + i * 256;
    xr[c] = te[c] + pe[c];
  }
}

// ---------------------------------------------------------------------------
// LayerNorm over C=1024, one block (256 thr) per row.
// ---------------------------------------------------------------------------
__global__ __launch_bounds__(256) void ln_kernel(
    const float* __restrict__ x, const float* __restrict__ w,
    const float* __restrict__ b, float* __restrict__ out) {
  __shared__ float red[256];
  const int row = blockIdx.x;
  const int t   = threadIdx.x;
  const float* xr = x + (size_t)row * C_n;
  float v0[4];
  float s = 0.f;
#pragma unroll
  for (int i = 0; i < 4; ++i) { v0[i] = xr[t + i * 256]; s += v0[i]; }
  red[t] = s; __syncthreads();
  for (int off = 128; off > 0; off >>= 1) {
    if (t < off) red[t] += red[t + off];
    __syncthreads();
  }
  const float mean = red[0] * (1.f / C_n);
  __syncthreads();
  float s2 = 0.f;
#pragma unroll
  for (int i = 0; i < 4; ++i) { const float d = v0[i] - mean; s2 += d * d; }
  red[t] = s2; __syncthreads();
  for (int off = 128; off > 0; off >>= 1) {
    if (t < off) red[t] += red[t + off];
    __syncthreads();
  }
  const float rstd = rsqrtf(red[0] * (1.f / C_n) + 1e-5f);
  float* outr = out + (size_t)row * C_n;
#pragma unroll
  for (int i = 0; i < 4; ++i) {
    const int c = t + i * 256;
    outr[c] = (v0[i] - mean) * rstd * w[c] + b[c];
  }
}

// ---------------------------------------------------------------------------
// WMMA GEMM: out[M=2048, N] = act( A[M,K] x W + bias ) + resid
// Weight addressing: W[k, n] at  k*wsk + (n>>6)*wsh + (n&63)
//   row-major (K x N)            -> wsk = N,  wsh = 64
//   head-blocked (H, C, HS=64)   -> wsk = 64, wsh = C*64
// Block = 256 threads (8 waves), 128x64 tile; wave w owns rows [16w,16w+16).
// Ash row-major (rows x k); Wsh TRANSPOSED (n x k) so every fragment load is
// a contiguous aligned 16B ds_load_b128. Double-buffered: stage chunk kk+32
// while WMMAs consume chunk kk; single barrier per chunk.
// ---------------------------------------------------------------------------
template <bool BIAS, bool RESID, bool GELU>
__global__ __launch_bounds__(256) void gemm_kernel(
    const float* __restrict__ A, const float* __restrict__ W,
    const float* __restrict__ bias, const float* __restrict__ resid,
    float* __restrict__ out, int N, int K, int wsk, int wsh) {
  __shared__ __align__(16) _Float16 Ash[2][128][40];  // 128 rows x 32 k
  __shared__ __align__(16) _Float16 Wsh[2][64][40];   // 64 n x 32 k (transp.)
  const int t    = threadIdx.x;
  const int lane = t & 31;
  const int wv   = t >> 5;
  const int hw   = lane >> 4;     // half-wave (0/1)
  const int ncol = lane & 15;
  const int m0 = blockIdx.y * 128;
  const int n0 = blockIdx.x * 64;

  AccF8 acc[4];
#pragma unroll
  for (int g = 0; g < 4; ++g)
#pragma unroll
    for (int r = 0; r < 8; ++r) acc[g].e[r] = 0.f;

  // A staging: thread -> (row = t>>1, 16 k's), contiguous 64B global reads.
  const int arow = t >> 1, ac0 = (t & 1) * 16;
  const float* aptr = A + (size_t)(m0 + arow) * K + ac0;
  // W staging (transposed): thread -> (n = t&63, 8 k's).
  // At fixed j, lanes read consecutive n -> coalesced global reads.
  const int wn = t & 63, wk0 = (t >> 6) * 8;
  const float* wptr = W + (size_t)(n0 >> 6) * (size_t)wsh + wn;

  auto stage = [&](int buf, int kk) {
    {   // A chunk (f32 -> f16), two aligned 16B LDS stores
      const float* p = aptr + kk;
      Frag8H t0, t1;
#pragma unroll
      for (int j = 0; j < 8; ++j) {
        t0.e[j] = (_Float16)p[j];
        t1.e[j] = (_Float16)p[8 + j];
      }
      *reinterpret_cast<v8h*>(&Ash[buf][arow][ac0])     = t0.v;
      *reinterpret_cast<v8h*>(&Ash[buf][arow][ac0 + 8]) = t1.v;
    }
    {   // W chunk transposed: Wsh[n][k], one aligned 16B LDS store
      Frag8H t0;
#pragma unroll
      for (int j = 0; j < 8; ++j)
        t0.e[j] = (_Float16)wptr[(size_t)(kk + wk0 + j) * wsk];
      *reinterpret_cast<v8h*>(&Wsh[buf][wn][wk0]) = t0.v;
    }
  };

  stage(0, 0);
  __syncthreads();

  for (int kk = 0; kk < K; kk += 32) {
    const int cur = (kk >> 5) & 1;
    if (kk + 32 < K) {
      // prefetch 2 chunks ahead on the activation stream (global_prefetch_b8)
      if (kk + 64 < K) __builtin_prefetch(aptr + kk + 64, 0, 1);
      stage(cur ^ 1, kk + 32);
    }

    // A fragment: lane holds row (lane&15), k = 8*hw + {0..7, 16..23}
    Frag16 fa;
    const int mrow = wv * 16 + ncol;
    const int kb8  = hw * 8;
    fa.h2[0] = *reinterpret_cast<const v8h*>(&Ash[cur][mrow][kb8]);
    fa.h2[1] = *reinterpret_cast<const v8h*>(&Ash[cur][mrow][kb8 + 16]);

    // B fragments: lane holds col (lane&15), k = 8*hw + {0..7, 16..23}
    const int kb16 = hw * 8;
#pragma unroll
    for (int g = 0; g < 4; ++g) {
      Frag16 fb;
      fb.h2[0] = *reinterpret_cast<const v8h*>(&Wsh[cur][g * 16 + ncol][kb16]);
      fb.h2[1] = *reinterpret_cast<const v8h*>(&Wsh[cur][g * 16 + ncol][kb16 + 16]);
      acc[g].v = wmma16(fa.v, fb.v, acc[g].v);
    }
    __syncthreads();
  }

  // Epilogue (fully specialized: no runtime feature branches).
#pragma unroll
  for (int g = 0; g < 4; ++g) {
#pragma unroll
    for (int r = 0; r < 8; ++r) {
      const int row = m0 + wv * 16 + r + hw * 8;
      const int n   = n0 + g * 16 + ncol;
      float val = acc[g].e[r];
      if (BIAS)  val += bias[n];
      if (GELU)  val = 0.5f * val * (1.f + erff(val * 0.70710678f));
      if (RESID) val += resid[(size_t)row * N + n];
      out[(size_t)row * N + n] = val;
    }
  }
}

// ---------------------------------------------------------------------------
// Flash attention (causal). One wave per (b, h, 16-row Q tile).
// S = (Q*scale) x K^T via WMMA, online softmax, O += P x V via WMMA.
// q/k/v/o layout: [B*T, C] with column = h*HS + d.
// V staged TRANSPOSED in LDS (d x key) so V B-fragments are vector loads.
// ---------------------------------------------------------------------------
__global__ __launch_bounds__(32) void attn_kernel(
    const float* __restrict__ qb, const float* __restrict__ kb,
    const float* __restrict__ vb, float* __restrict__ ob) {
  __shared__ __align__(16) _Float16 Vsh[64][40];  // 64 dims x 32 keys (transp.)
  __shared__ __align__(16) _Float16 Psh[16][40];  // 16 rows x 32 keys
  const int lane = threadIdx.x;
  const int q0   = blockIdx.x * 16;
  const int hh   = blockIdx.y;
  const int bb   = blockIdx.z;
  const int hw   = lane >> 4;
  const int ncol = lane & 15;
  const size_t bhbase = (size_t)bb * T_n * C_n + (size_t)hh * HS_n;

  // Q A-fragments for d-chunks 0..31 and 32..63, pre-scaled by HS^-0.5
  Frag16 qa[2];
  {
    const float* qrow = qb + bhbase + (size_t)(q0 + ncol) * C_n + hw * 16;
#pragma unroll
    for (int c = 0; c < 2; ++c) {
#pragma unroll
      for (int e = 0; e < 8; ++e)
        qa[c].e[e] = (_Float16)(qrow[32 * c + e] * 0.125f);
#pragma unroll
      for (int e = 0; e < 8; ++e)
        qa[c].e[8 + e] = (_Float16)(qrow[32 * c + 8 + e] * 0.125f);
    }
  }

  AccF8 oacc[4];
  float mrun[8], lrun[8];
#pragma unroll
  for (int r = 0; r < 8; ++r) { mrun[r] = -INFINITY; lrun[r] = 0.f; }
#pragma unroll
  for (int g = 0; g < 4; ++g)
#pragma unroll
    for (int r = 0; r < 8; ++r) oacc[g].e[r] = 0.f;

  const int ktmax = (q0 + 15) >> 5;
  for (int kt = 0; kt <= ktmax; ++kt) {
    const int kbase = kt * 32;

    // Stage V tile transposed: lane = key, contiguous global reads,
    // scalar transposed LDS stores (vector read-side fragment loads later).
    {
      const float* vrow = vb + bhbase + (size_t)(kbase + lane) * C_n;
#pragma unroll
      for (int d0 = 0; d0 < 64; ++d0)
        Vsh[d0][lane] = (_Float16)vrow[d0];
    }

    // K^T B-fragments straight from global (contiguous 64B per lane)
    Frag16 kf0[2], kf1[2];
#pragma unroll
    for (int c = 0; c < 2; ++c) {
      const float* kp0 =
          kb + bhbase + (size_t)(kbase + ncol) * C_n + 32 * c + 16 * hw;
      const float* kp1 =
          kb + bhbase + (size_t)(kbase + 16 + ncol) * C_n + 32 * c + 16 * hw;
#pragma unroll
      for (int e = 0; e < 16; ++e) {
        kf0[c].e[e] = (_Float16)kp0[e];
        kf1[c].e[e] = (_Float16)kp1[e];
      }
    }

    AccF8 s0, s1;
#pragma unroll
    for (int r = 0; r < 8; ++r) { s0.e[r] = 0.f; s1.e[r] = 0.f; }
    s0.v = wmma16(qa[0].v, kf0[0].v, s0.v);
    s0.v = wmma16(qa[1].v, kf0[1].v, s0.v);
    s1.v = wmma16(qa[0].v, kf1[0].v, s1.v);
    s1.v = wmma16(qa[1].v, kf1[1].v, s1.v);

    // Online softmax; row m = r + 8*hw lives entirely in one half-wave.
#pragma unroll
    for (int r = 0; r < 8; ++r) {
      const int m  = r + hw * 8;
      const int qg = q0 + m;
      const int j0 = kbase + ncol;
      const int j1 = j0 + 16;
      float a0 = (j0 <= qg) ? s0.e[r] : -INFINITY;
      float a1 = (j1 <= qg) ? s1.e[r] : -INFINITY;
      float tm = fmaxf(a0, a1);
      tm = fmaxf(tm, __shfl_xor(tm, 1));
      tm = fmaxf(tm, __shfl_xor(tm, 2));
      tm = fmaxf(tm, __shfl_xor(tm, 4));
      tm = fmaxf(tm, __shfl_xor(tm, 8));
      const float mnew  = fmaxf(mrun[r], tm);
      const float alpha = __expf(mrun[r] - mnew);
      const float p0 = __expf(a0 - mnew);
      const float p1 = __expf(a1 - mnew);
      float rs = p0 + p1;
      rs += __shfl_xor(rs, 1);
      rs += __shfl_xor(rs, 2);
      rs += __shfl_xor(rs, 4);
      rs += __shfl_xor(rs, 8);
      lrun[r] = lrun[r] * alpha + rs;
      mrun[r] = mnew;
#pragma unroll
      for (int g = 0; g < 4; ++g) oacc[g].e[r] *= alpha;
      Psh[m][ncol]      = (_Float16)p0;
      Psh[m][ncol + 16] = (_Float16)p1;
    }
    __syncthreads();

    // P as A-fragment (C/D -> A layout via LDS), vector loads
    Frag16 pa;
    pa.h2[0] = *reinterpret_cast<const v8h*>(&Psh[ncol][hw * 8]);
    pa.h2[1] = *reinterpret_cast<const v8h*>(&Psh[ncol][hw * 8 + 16]);

    // V B-fragments from transposed LDS: k = 8*hw + {0..7, 16..23}
#pragma unroll
    for (int g = 0; g < 4; ++g) {
      Frag16 fv;
      fv.h2[0] = *reinterpret_cast<const v8h*>(&Vsh[g * 16 + ncol][hw * 8]);
      fv.h2[1] = *reinterpret_cast<const v8h*>(&Vsh[g * 16 + ncol][hw * 8 + 16]);
      oacc[g].v = wmma16(pa.v, fv.v, oacc[g].v);
    }
    __syncthreads();
  }

  // Normalize and write O
#pragma unroll
  for (int g = 0; g < 4; ++g) {
#pragma unroll
    for (int r = 0; r < 8; ++r) {
      const int m = r + hw * 8;
      ob[bhbase + (size_t)(q0 + m) * C_n + g * 16 + ncol] =
          oacc[g].e[r] / lrun[r];
    }
  }
}

// ---------------------------------------------------------------------------
// Host orchestration
// ---------------------------------------------------------------------------
static void run_gemm(const float* A, const float* W, const float* bias,
                     const float* resid, float* out, int N, int K, int wsk,
                     int wsh, int act, hipStream_t s) {
  dim3 grid(N / 64, M_n / 128), block(256);
  if (bias && resid)
    gemm_kernel<true, true, false><<<grid, block, 0, s>>>(A, W, bias, resid, out, N, K, wsk, wsh);
  else if (bias && act)
    gemm_kernel<true, false, true><<<grid, block, 0, s>>>(A, W, bias, resid, out, N, K, wsk, wsh);
  else if (bias)
    gemm_kernel<true, false, false><<<grid, block, 0, s>>>(A, W, bias, resid, out, N, K, wsk, wsh);
  else
    gemm_kernel<false, false, false><<<grid, block, 0, s>>>(A, W, bias, resid, out, N, K, wsk, wsh);
}

extern "C" void kernel_launch(void* const* d_in, const int* in_sizes, int n_in,
                              void* d_out, int out_size, void* d_ws,
                              size_t ws_size, hipStream_t stream) {
  (void)in_sizes; (void)n_in; (void)out_size; (void)ws_size;
  const int*   idx   = (const int*)  d_in[0];
  const float* tok   = (const float*)d_in[1];
  const float* pos   = (const float*)d_in[2];
  const float* ln1w  = (const float*)d_in[3];
  const float* ln1b  = (const float*)d_in[4];
  const float* wq    = (const float*)d_in[5];
  const float* wk    = (const float*)d_in[6];
  const float* wv    = (const float*)d_in[7];
  const float* projw = (const float*)d_in[8];
  const float* projb = (const float*)d_in[9];
  const float* ln2w  = (const float*)d_in[10];
  const float* ln2b  = (const float*)d_in[11];
  const float* w1    = (const float*)d_in[12];
  const float* b1    = (const float*)d_in[13];
  const float* w2    = (const float*)d_in[14];
  const float* b2    = (const float*)d_in[15];
  const float* lnfw  = (const float*)d_in[16];
  const float* lnfb  = (const float*)d_in[17];
  const float* lmw   = (const float*)d_in[18];
  const float* lmb   = (const float*)d_in[19];

  const size_t MC = (size_t)M_n * C_n;
  float* x = (float*)d_ws;
  float* h = x + MC;
  float* q = h + MC;
  float* k = q + MC;
  float* v = k + MC;
  float* o = v + MC;
  float* g = o + MC;           // M x 4C hidden

  embed_kernel<<<dim3(M_n), dim3(256), 0, stream>>>(idx, tok, pos, x);

  for (int l = 0; l < L_n; ++l) {
    const size_t co  = (size_t)l * C_n;
    const size_t hco = (size_t)l * H_n * C_n * HS_n;     // = l*C*C
    const size_t cco = (size_t)l * C_n * C_n;
    const size_t c4o = (size_t)l * C_n * 4 * C_n;
    const size_t f4o = (size_t)l * 4 * C_n;

    ln_kernel<<<dim3(M_n), dim3(256), 0, stream>>>(x, ln1w + co, ln1b + co, h);
    // QKV: weights are (H, C, HS) blocks -> wsk=HS, wsh=C*HS
    run_gemm(h, wq + hco, nullptr, nullptr, q, C_n, C_n, HS_n, C_n * HS_n, 0, stream);
    run_gemm(h, wk + hco, nullptr, nullptr, k, C_n, C_n, HS_n, C_n * HS_n, 0, stream);
    run_gemm(h, wv + hco, nullptr, nullptr, v, C_n, C_n, HS_n, C_n * HS_n, 0, stream);

    attn_kernel<<<dim3(T_n / 16, H_n, B_n), dim3(32), 0, stream>>>(q, k, v, o);

    // proj + bias + residual (in-place on x is safe: 1 read/write per elem)
    run_gemm(o, projw + cco, projb + co, x, x, C_n, C_n, C_n, 64, 0, stream);

    ln_kernel<<<dim3(M_n), dim3(256), 0, stream>>>(x, ln2w + co, ln2b + co, h);
    run_gemm(h, w1 + c4o, b1 + f4o, nullptr, g, 4 * C_n, C_n, 4 * C_n, 64, 1, stream);
    run_gemm(g, w2 + c4o, b2 + co, x, x, C_n, 4 * C_n, C_n, 64, 0, stream);
  }

  ln_kernel<<<dim3(M_n), dim3(256), 0, stream>>>(x, lnfw, lnfb, h);
  run_gemm(h, lmw, lmb, nullptr, (float*)d_out, V_n, C_n, V_n, 64, 0, stream);
}
